// CityModel_57088705298521
// MI455X (gfx1250) — matrix operand order repaired and checked by reading, not compiled, not run on hardware
//
#include <hip/hip_runtime.h>
#include <hip/hip_bf16.h>

// ---------------- model constants ----------------
#define BB    8
#define SS    512
#define TT    24
#define PP    24
#define EE    8192
#define NN    (BB * SS * TT)      // 98304 nodes
#define MM    (TT * BB * EE)      // 1572864 edges
#define GNN_H 32
#define RH    128                 // RNN hidden
#define GATES (4 * RH)            // 512

typedef __attribute__((ext_vector_type(16))) __bf16 v16bf;
typedef __attribute__((ext_vector_type(8)))  float  v8f;

union Frag16 {
    v16bf  v;
    __bf16 h[16];
    uint4  q[2];
};

__device__ __forceinline__ float sigmf(float x) { return 1.0f / (1.0f + __expf(-x)); }

// ================= kernel 1: edge messages + scatter-mean accumulation ==========
__global__ __launch_bounds__(256) void edge_kernel(
    const float* __restrict__ aqi,      // [N] flat
    const int*   __restrict__ conn,     // [B,E,2]
    const float* __restrict__ wattr,    // [B*T*E, 2] flat
    const float* __restrict__ W1,       // [4,32]
    const float* __restrict__ b1,       // [32]
    float* __restrict__ agg,            // [N,32]
    float* __restrict__ cnt)            // [N]
{
    __shared__ float sW1[4 * 32];
    __shared__ float sb1[32];
    if (threadIdx.x < 128) sW1[threadIdx.x] = W1[threadIdx.x];
    if (threadIdx.x < 32)  sb1[threadIdx.x] = b1[threadIdx.x];
    __syncthreads();

    size_t k = (size_t)blockIdx.x * 256 + threadIdx.x;   // one thread per edge
    int i  = (int)(k >> 13);          // k / E
    int e  = (int)(k & (EE - 1));     // k % E
    int bb = i & (BB - 1);            // i % B  (conn tiled T times over batch)
    int off = i * TT;                 // faithful to reference: arange(T*B)*T offset
    int rown = conn[((size_t)bb * EE + e) * 2 + 0] + off;
    int coln = conn[((size_t)bb * EE + e) * 2 + 1] + off;

    float xr = aqi[rown];
    float xc = aqi[coln];
    float w0 = wattr[k * 2 + 0];
    float w1 = wattr[k * 2 + 1];

    #pragma unroll 4
    for (int j = 0; j < 32; ++j) {
        float m = sb1[j] + xr * sW1[j] + xc * sW1[32 + j] + w0 * sW1[64 + j] + w1 * sW1[96 + j];
        atomicAdd(&agg[(size_t)coln * 32 + j], m);
    }
    atomicAdd(&cnt[coln], 1.0f);
}

// ================= kernel 2: node update -> g (bf16 row-major [N,32]) ===========
__global__ __launch_bounds__(256) void node_kernel(
    const float* __restrict__ aqi,      // [N]
    const float* __restrict__ agg,      // [N,32]
    const float* __restrict__ cnt,      // [N]
    const float* __restrict__ wea,      // [B*T, 11] flat
    const float* __restrict__ W2,       // [44,32]
    const float* __restrict__ b2,       // [32]
    __bf16* __restrict__ gbf)           // [N,32] bf16
{
    __shared__ float sW2[44 * 32];
    __shared__ float sb2[32];
    for (int idx = threadIdx.x; idx < 44 * 32; idx += 256) sW2[idx] = W2[idx];
    if (threadIdx.x < 32) sb2[threadIdx.x] = b2[threadIdx.x];
    __syncthreads();

    int n = blockIdx.x * 256 + threadIdx.x;
    float in[44];
    in[0] = aqi[n];
    float inv = 1.0f / fmaxf(cnt[n], 1.0f);
    #pragma unroll 8
    for (int j = 0; j < 32; ++j) in[1 + j] = agg[(size_t)n * 32 + j] * inv;
    const float* up = wea + (size_t)(n % (BB * TT)) * 11;   // u = tile over S
    #pragma unroll
    for (int j = 0; j < 11; ++j) in[33 + j] = up[j];

    for (int j = 0; j < 32; ++j) {
        float s = sb2[j];
        #pragma unroll 4
        for (int i2 = 0; i2 < 44; ++i2) s += in[i2] * sW2[i2 * 32 + j];
        gbf[(size_t)n * 32 + j] = (__bf16)s;
    }
}

// ================= kernel 3: weight repack into WMMA B-fragment order ===========
// B-fragment (16-bit, 32x16): lane holds col N=lane%16, K base = 8*(lane/16);
// element j = W[col][kc*32 + (lane/16)*8 + (j%8) + (j/8)*16].  Storage:
//   frag[ ((gt*KC + kc)*32 + lane)*16 + j ]
__device__ __forceinline__ void pack_one(const float* W, __bf16* F, int KC, int Kreal, int e2)
{
    int per  = KC * 512;
    int gt   = e2 / per;
    int rem  = e2 % per;
    int kc   = rem / 512;
    int l2   = rem % 512;
    int lane = l2 / 16;
    int j    = l2 % 16;
    int col  = gt * 16 + (lane & 15);
    int k    = kc * 32 + (lane >> 4) * 8 + (j & 7) + (j >> 3) * 16;
    F[e2] = (k < Kreal) ? (__bf16)W[(size_t)col * Kreal + k] : (__bf16)0.0f;
}

__global__ __launch_bounds__(256) void prep_weights(
    const float* __restrict__ WihE, const float* __restrict__ WhhE,
    const float* __restrict__ WihD, const float* __restrict__ WhhD,
    const float* __restrict__ bihE, const float* __restrict__ bhhE,
    const float* __restrict__ bihD, const float* __restrict__ bhhD,
    __bf16* fWihE, __bf16* fWhhE, __bf16* fWihD, __bf16* fWhhD,
    float* biasE, float* biasD)
{
    int idx = blockIdx.x * 256 + threadIdx.x;
    if (idx < GATES) {
        biasE[idx] = bihE[idx] + bhhE[idx];
        biasD[idx] = bihD[idx] + bhhD[idx];
    }
    if (idx < 16384)                      pack_one(WihE, fWihE, 1, 32,  idx);
    else if (idx < 16384 + 65536)         pack_one(WhhE, fWhhE, 4, 128, idx - 16384);
    else if (idx < 16384 + 65536 + 16384) pack_one(WihD, fWihD, 1, 12, idx - 81920);
    else if (idx < 163840)                pack_one(WhhD, fWhhD, 4, 128, idx - 98304);
}

// ================= fused LSTM encoder + autoregressive decoder =================
// One wave owns 16 batch rows for all 48 steps.  h (bf16) and c (f32) live in a
// private per-wave LDS region; WMMA operands stream weights from L2.

__device__ __forceinline__ void load_h_frags(const __bf16* myH, int l, int hi, v16bf ah[4])
{
    #pragma unroll
    for (int kc = 0; kc < 4; ++kc) {
        Frag16 f;
        const __bf16* hp = myH + l * RH + kc * 32 + hi * 8;
        f.q[0] = *(const uint4*)hp;          // K: kc*32 + hi*8 .. +7
        f.q[1] = *(const uint4*)(hp + 16);   // K: kc*32 + 16 + hi*8 .. +7
        ah[kc] = f.v;
    }
}

__device__ __forceinline__ void lstm_step(
    v16bf ax, const v16bf ah[4],
    const __bf16* __restrict__ fX, const __bf16* __restrict__ fH,
    const float* __restrict__ bias,
    __bf16* myH, float* myC, int lane, int l, int hi)
{
    // NOTE: nt loop intentionally NOT unrolled: keeps the 20 weight-fragment
    // loads per iteration loop-carried so they cannot be hoisted out of the
    // time-step loop and spilled to scratch (round-1 failure mode).
    #pragma unroll 1
    for (int nt = 0; nt < 8; ++nt) {
        v8f acc[4];
        #pragma unroll
        for (int q = 0; q < 4; ++q) {
            int gt = q * 8 + nt;                       // gate col tile in [0,32)
            float bv = bias[gt * 16 + l];              // D-frag col = lane%16
            v8f a;
            #pragma unroll
            for (int z = 0; z < 8; ++z) a[z] = bv;
            // input projection: K = 32 (1 fragment)
            {
                const v16bf* bp = (const v16bf*)(fX + ((size_t)gt * 32 + lane) * 16);
                a = __builtin_amdgcn_wmma_f32_16x16x32_bf16(
                        false, ax, false, *bp, (short)0, a, false, false);
            }
            // recurrent projection: K = 128 (4 fragments)
            #pragma unroll
            for (int kc = 0; kc < 4; ++kc) {
                const v16bf* bp = (const v16bf*)(fH + (((size_t)gt * 4 + kc) * 32 + lane) * 16);
                a = __builtin_amdgcn_wmma_f32_16x16x32_bf16(
                        false, ah[kc], false, *bp, (short)0, a, false, false);
            }
            acc[q] = a;
        }
        // gate order i, f, g, o  (torch LSTM); c in LDS (f32), h out as bf16
        #pragma unroll
        for (int vv = 0; vv < 8; ++vv) {
            int eidx = (vv + 8 * hi) * RH + nt * 16 + l;   // (row, col) in 16x128
            float iv = sigmf(acc[0][vv]);
            float fv = sigmf(acc[1][vv]);
            float gv = tanhf(acc[2][vv]);
            float ov = sigmf(acc[3][vv]);
            float cn = fv * myC[eidx] + iv * gv;
            myC[eidx] = cn;
            myH[eidx] = (__bf16)(ov * tanhf(cn));
        }
    }
}

__global__ __launch_bounds__(128) void lstm_enc_dec(
    const __bf16* __restrict__ gbf,     // [N,32] bf16 (encoder input, flat node idx)
    const float*  __restrict__ h0,      // [4096,128]
    const float*  __restrict__ c0,      // [4096,128]
    const __bf16* __restrict__ fWihE, const __bf16* __restrict__ fWhhE,
    const __bf16* __restrict__ fWihD, const __bf16* __restrict__ fWhhD,
    const float*  __restrict__ biasE, const float* __restrict__ biasD,
    const float*  __restrict__ aqi,     // [N] flat (for decoder seed)
    const float*  __restrict__ sta_for, // [B,P,11]
    const float*  __restrict__ Wl,      // [128]
    const float*  __restrict__ bl,      // [1]
    float* __restrict__ outp)           // [B*S, P]
{
    __shared__ __attribute__((aligned(32))) __bf16 ldsH[4][16 * RH];   // 16 KB
    __shared__ float ldsC[4][16 * RH];                                  // 32 KB

    int lane = threadIdx.x & 31;
    int wave = threadIdx.x >> 5;
    int l  = lane & 15;
    int hi = lane >> 4;
    int r0 = (blockIdx.x * 4 + wave) * 16;   // first batch row of this wave's tile
    int b  = r0 / SS;
    int s0 = r0 % SS;
    __bf16* myH = ldsH[wave];
    float*  myC = ldsC[wave];

    // ---- init h (bf16) and c (f32) in this wave's LDS region ----
    for (int e = lane; e < 16 * RH; e += 32) {
        int row = e / RH, col = e % RH;
        myH[e] = (__bf16)h0[(size_t)(r0 + row) * RH + col];
        myC[e] = c0[(size_t)(r0 + row) * RH + col];
    }

    v16bf ah[4];
    load_h_frags(myH, l, hi, ah);

    // ---- encoder: enc_in[r, t] = g[(b*T + t)*S + s]  (faithful reshape quirk) ----
    for (int t = 0; t < TT; ++t) {
        Frag16 fx;
        const __bf16* gp = gbf + ((size_t)(b * TT + t) * SS + s0 + l) * 32;
        fx.q[0] = *(const uint4*)(gp + hi * 8);
        fx.q[1] = *(const uint4*)(gp + 16 + hi * 8);
        lstm_step(fx.v, ah, fWihE, fWhhE, biasE, myH, myC, lane, l, hi);
        load_h_frags(myH, l, hi, ah);
    }

    // ---- autoregressive decoder ----
    // seed: last observed aqi for this row
    float outv = aqi[(size_t)(b * SS + s0 + l) * TT + (TT - 1)];

    for (int p = 0; p < PP; ++p) {
        // decoder input xt = [prev_out, sta_for[r % B][p][0..10]]  (K padded to 32)
        Frag16 fx;
        #pragma unroll
        for (int j = 0; j < 16; ++j) fx.h[j] = (__bf16)0.0f;
        const float* fp = sta_for + ((size_t)((r0 + l) & (BB - 1)) * PP + p) * 11;
        if (hi == 0) {
            fx.h[0] = (__bf16)outv;                        // K0
            #pragma unroll
            for (int j = 1; j < 8; ++j) fx.h[j] = (__bf16)fp[j - 1];   // K1..7
        } else {
            #pragma unroll
            for (int j = 0; j < 4; ++j) fx.h[j] = (__bf16)fp[7 + j];   // K8..11
        }
        lstm_step(fx.v, ah, fWihD, fWhhD, biasD, myH, myC, lane, l, hi);
        load_h_frags(myH, l, hi, ah);

        // readout: out = h . Wl + bl   (each lane computes its row; lanes>=16 duplicate)
        float s = bl[0];
        #pragma unroll 8
        for (int k = 0; k < RH; ++k) s += (float)myH[l * RH + k] * Wl[k];
        outv = s;
        if (hi == 0) outp[(size_t)(r0 + l) * PP + p] = s;
    }
}

// ================================ host launcher ================================
extern "C" void kernel_launch(void* const* d_in, const int* in_sizes, int n_in,
                              void* d_out, int out_size, void* d_ws, size_t ws_size,
                              hipStream_t stream)
{
    const float* sta_aqi  = (const float*)d_in[0];
    const int*   sta_conn = (const int*)  d_in[1];
    const float* sta_w    = (const float*)d_in[2];
    const float* sta_wea  = (const float*)d_in[3];
    const float* sta_for  = (const float*)d_in[4];
    // d_in[5] city_u: unused by the reference
    const float* h0   = (const float*)d_in[6];
    const float* c0   = (const float*)d_in[7];
    const float* W1   = (const float*)d_in[8];
    const float* b1   = (const float*)d_in[9];
    const float* W2   = (const float*)d_in[10];
    const float* b2   = (const float*)d_in[11];
    const float* WihE = (const float*)d_in[12];
    const float* WhhE = (const float*)d_in[13];
    const float* bihE = (const float*)d_in[14];
    const float* bhhE = (const float*)d_in[15];
    const float* WihD = (const float*)d_in[16];
    const float* WhhD = (const float*)d_in[17];
    const float* bihD = (const float*)d_in[18];
    const float* bhhD = (const float*)d_in[19];
    const float* Wl   = (const float*)d_in[20];
    const float* bl   = (const float*)d_in[21];

    char* ws = (char*)d_ws;
    size_t off = 0;
    auto carve = [&](size_t bytes) -> void* {
        void* p = ws + off;
        off += (bytes + 255) & ~(size_t)255;
        return p;
    };
    float*  agg   = (float*) carve((size_t)NN * 32 * sizeof(float));
    float*  cnt   = (float*) carve((size_t)NN * sizeof(float));
    __bf16* gbf   = (__bf16*)carve((size_t)NN * 32 * sizeof(__bf16));
    __bf16* fWihE = (__bf16*)carve((size_t)16384 * sizeof(__bf16));
    __bf16* fWhhE = (__bf16*)carve((size_t)65536 * sizeof(__bf16));
    __bf16* fWihD = (__bf16*)carve((size_t)16384 * sizeof(__bf16));
    __bf16* fWhhD = (__bf16*)carve((size_t)65536 * sizeof(__bf16));
    float*  biasE = (float*) carve((size_t)GATES * sizeof(float));
    float*  biasD = (float*) carve((size_t)GATES * sizeof(float));

    hipMemsetAsync(agg, 0, (size_t)NN * 32 * sizeof(float), stream);
    hipMemsetAsync(cnt, 0, (size_t)NN * sizeof(float), stream);

    prep_weights<<<640, 256, 0, stream>>>(WihE, WhhE, WihD, WhhD,
                                          bihE, bhhE, bihD, bhhD,
                                          fWihE, fWhhE, fWihD, fWhhD,
                                          biasE, biasD);

    edge_kernel<<<MM / 256, 256, 0, stream>>>(sta_aqi, sta_conn, sta_w, W1, b1, agg, cnt);

    node_kernel<<<NN / 256, 256, 0, stream>>>(sta_aqi, agg, cnt, sta_wea, W2, b2, gbf);

    lstm_enc_dec<<<64, 128, 0, stream>>>(gbf, h0, c0,
                                         fWihE, fWhhE, fWihD, fWhhD,
                                         biasE, biasD,
                                         sta_aqi, sta_for, Wl, bl,
                                         (float*)d_out);
}
